// SelectiveSSM_68771016343889
// MI455X (gfx1250) — matrix operand order repaired and checked by reading, not compiled, not run on hardware
//
#include <hip/hip_runtime.h>
#include <hip/hip_bf16.h>

// ---------------------------------------------------------------------------
// Problem constants (fixed by the reference): BSZ=2, L=2048, D=1024, N=16
// ---------------------------------------------------------------------------
#define BSZ   2
#define SEQL  2048
#define DIM   1024
#define NST   16
#define MROWS (BSZ * SEQL)          // 4096 GEMM rows
#define KDIM  DIM                   // 1024 reduction dim

// GEMM tiling
#define MT 128
#define NT 128
#define KT 32
#define LDT (KT + 8)                // padded LDS row stride (bf16 elems, 80B rows -> 16B aligned chunks)

typedef __attribute__((ext_vector_type(4)))  __bf16 v4bf;
typedef __attribute__((ext_vector_type(8)))  __bf16 v8bf;
typedef __attribute__((ext_vector_type(16))) __bf16 v16bf;
typedef __attribute__((ext_vector_type(8)))  float  v8f;

// ---------------------------------------------------------------------------
// LDS -> WMMA fragment loaders (CDNA5 wave32 16-bit layouts, ISA 7.12.2)
// A-matrix 16x32: lane m = lane%16; elems 0..7  -> K = (lane/16)*8 + 0..7
//                                   elems 8..15 -> K = 16 + (lane/16)*8 + 0..7
// ---------------------------------------------------------------------------
__device__ __forceinline__ v16bf load_a_frag(const __bf16* lds, int row0, int lane) {
    const __bf16* rp = lds + (row0 + (lane & 15)) * LDT;
    int half8 = (lane >> 4) * 8;
    v8bf x0 = *(const v8bf*)(rp + half8);
    v8bf x1 = *(const v8bf*)(rp + 16 + half8);
    v16bf r;
#pragma unroll
    for (int i = 0; i < 8; ++i) { r[i] = x0[i]; r[8 + i] = x1[i]; }
    return r;
}

// B-matrix 32x16 (we feed W rows as B columns since Y = X * W^T):
// lane n = lane%16; elems 0..7 -> K = (lane/16)*8 + 0..7, elems 8..15 -> K = 16 + (lane/16)*8 + 0..7
__device__ __forceinline__ v16bf load_b_frag(const __bf16* lds, int row0, int lane) {
    const __bf16* rp = lds + (row0 + (lane & 15)) * LDT + (lane >> 4) * 8;
    v8bf x0 = *(const v8bf*)(rp);
    v8bf x1 = *(const v8bf*)(rp + 16);
    v16bf r;
#pragma unroll
    for (int i = 0; i < 8; ++i) { r[i] = x0[i]; r[8 + i] = x1[i]; }
    return r;
}

__device__ __forceinline__ __bf16 bf_hi(float v) { return (__bf16)v; }
__device__ __forceinline__ __bf16 bf_lo(float v, __bf16 hi) { return (__bf16)(v - (float)hi); }

// ---------------------------------------------------------------------------
// Fused projection GEMM: Y(M,E) = X(M,K) * W(E,K)^T  [+bias, +softplus]
// fp32 inputs, split-bf16 (hi/lo) WMMA with 3-term compensation.
// Block: 256 threads (8 waves), 128x128 tile, wave -> 64x32 subtile.
// ---------------------------------------------------------------------------
template <bool SOFTPLUS>
__global__ __launch_bounds__(256)
void ssm_proj_gemm(const float* __restrict__ X, const float* __restrict__ W,
                   const float* __restrict__ bias, float* __restrict__ Y, int E) {
    __shared__ __bf16 sA_hi[MT * LDT];
    __shared__ __bf16 sA_lo[MT * LDT];
    __shared__ __bf16 sB_hi[NT * LDT];
    __shared__ __bf16 sB_lo[NT * LDT];

    const int tid    = threadIdx.x;
    const int lane   = tid & 31;
    const int waveId = tid >> 5;
    const int wave_m = waveId >> 2;      // 0..1  -> 64-row slab
    const int wave_n = waveId & 3;       // 0..3  -> 32-col slab
    const int gm0    = blockIdx.y * MT;
    const int gn0    = blockIdx.x * NT;

    v8f acc[4][2];
#pragma unroll
    for (int mi = 0; mi < 4; ++mi)
#pragma unroll
        for (int ni = 0; ni < 2; ++ni) acc[mi][ni] = (v8f)(0.0f);

    for (int k0 = 0; k0 < KDIM; k0 += KT) {
        __syncthreads();
        // Stage X and W tiles: 128 rows x 32 f32 each; float4 per thread x4.
#pragma unroll
        for (int it = 0; it < 4; ++it) {
            int ci = tid + it * 256;
            int r  = ci >> 3;
            int c  = (ci & 7) << 2;
            float4 vx = *(const float4*)(X + (size_t)(gm0 + r) * KDIM + k0 + c);
            float4 vw = *(const float4*)(W + (size_t)(gn0 + r) * KDIM + k0 + c);
            float fx[4] = {vx.x, vx.y, vx.z, vx.w};
            float fw[4] = {vw.x, vw.y, vw.z, vw.w};
            v4bf xh, xl, wh, wl;
#pragma unroll
            for (int j = 0; j < 4; ++j) {
                __bf16 h;
                h = bf_hi(fx[j]); xh[j] = h; xl[j] = bf_lo(fx[j], h);
                h = bf_hi(fw[j]); wh[j] = h; wl[j] = bf_lo(fw[j], h);
            }
            *(v4bf*)&sA_hi[r * LDT + c] = xh;
            *(v4bf*)&sA_lo[r * LDT + c] = xl;
            *(v4bf*)&sB_hi[r * LDT + c] = wh;
            *(v4bf*)&sB_lo[r * LDT + c] = wl;
        }
        __syncthreads();

        v16bf a_hi[4], a_lo[4], b_hi[2], b_lo[2];
#pragma unroll
        for (int mi = 0; mi < 4; ++mi) {
            int r0 = wave_m * 64 + mi * 16;
            a_hi[mi] = load_a_frag(sA_hi, r0, lane);
            a_lo[mi] = load_a_frag(sA_lo, r0, lane);
        }
#pragma unroll
        for (int ni = 0; ni < 2; ++ni) {
            int r0 = wave_n * 32 + ni * 16;
            b_hi[ni] = load_b_frag(sB_hi, r0, lane);
            b_lo[ni] = load_b_frag(sB_lo, r0, lane);
        }
#pragma unroll
        for (int mi = 0; mi < 4; ++mi)
#pragma unroll
            for (int ni = 0; ni < 2; ++ni) {
                acc[mi][ni] = __builtin_amdgcn_wmma_f32_16x16x32_bf16(
                    false, a_hi[mi], false, b_hi[ni], (short)0, acc[mi][ni], false, false);
                acc[mi][ni] = __builtin_amdgcn_wmma_f32_16x16x32_bf16(
                    false, a_hi[mi], false, b_lo[ni], (short)0, acc[mi][ni], false, false);
                acc[mi][ni] = __builtin_amdgcn_wmma_f32_16x16x32_bf16(
                    false, a_lo[mi], false, b_hi[ni], (short)0, acc[mi][ni], false, false);
            }
    }

    // Epilogue. C/D layout: VGPR i -> M = i + 8*(lane>=16), N = lane%16.
    const int half = lane >> 4;
    const int ncol = lane & 15;
#pragma unroll
    for (int mi = 0; mi < 4; ++mi) {
#pragma unroll
        for (int ni = 0; ni < 2; ++ni) {
            int gm_base = gm0 + wave_m * 64 + mi * 16 + half * 8;
            int gn      = gn0 + wave_n * 32 + ni * 16 + ncol;
            float b = 0.0f;
            if (SOFTPLUS) b = bias[gn];
#pragma unroll
            for (int r = 0; r < 8; ++r) {
                float v = acc[mi][ni][r];
                if (SOFTPLUS) {
                    v += b;
                    v = (v > 20.0f) ? v : log1pf(__expf(v));  // softplus
                }
                Y[(size_t)(gm_base + r) * E + gn] = v;
            }
        }
    }
}

// ---------------------------------------------------------------------------
// Selective scan: one lane per (b, d, n) state; sequential over L.
// h_{t} = exp(dt*A) h_{t-1} + ((exp(dt*A)-1)/A) * B_t * x_t ;  y = sum_n h*C
// ---------------------------------------------------------------------------
__global__ __launch_bounds__(256)
void ssm_scan(const float* __restrict__ x, const float* __restrict__ delta,
              const float* __restrict__ Bsel, const float* __restrict__ Csel,
              const float* __restrict__ A, float* __restrict__ y) {
    int t = blockIdx.x * blockDim.x + threadIdx.x;   // 0 .. BSZ*DIM*NST-1
    int n = t & (NST - 1);
    int p = t >> 4;                                   // (b,d) pair
    int d = p & (DIM - 1);
    int b = p >> 10;
    if (b >= BSZ) return;

    const float a     = A[d * NST + n];
    const float inv_a = 1.0f / a;
    float h = 0.0f;

    for (int l = 0; l < SEQL; ++l) {
        size_t base = ((size_t)b * SEQL + l) * DIM + d;
        float xv = x[base];
        float dt = delta[base];
        float Bv = Bsel[base * NST + n];
        float Cv = Csel[base * NST + n];

        float Abar = __expf(dt * a);                 // ZOH A
        float Bbar = (Abar - 1.0f) * inv_a * Bv;     // exact ZOH B
        h = Abar * h + Bbar * xv;

        float c = h * Cv;
        // reduce over the 16 n-lanes (stays inside the 16-group for masks <=8)
        c += __shfl_xor(c, 1);
        c += __shfl_xor(c, 2);
        c += __shfl_xor(c, 4);
        c += __shfl_xor(c, 8);
        if (n == 0) y[base] = c;
    }
}

// ---------------------------------------------------------------------------
// Launcher
// ---------------------------------------------------------------------------
extern "C" void kernel_launch(void* const* d_in, const int* in_sizes, int n_in,
                              void* d_out, int out_size, void* d_ws, size_t ws_size,
                              hipStream_t stream) {
    const float* x        = (const float*)d_in[0];   // (2,2048,1024)
    const float* A        = (const float*)d_in[1];   // (1024,16)
    const float* W_delta  = (const float*)d_in[2];   // (1024,1024)
    const float* d_bias   = (const float*)d_in[3];   // (1024,)
    const float* W_B      = (const float*)d_in[4];   // (16384,1024)
    const float* W_C      = (const float*)d_in[5];   // (16384,1024)
    float* y = (float*)d_out;

    char* ws = (char*)d_ws;
    float* delta = (float*)(ws);                                          // 16 MiB
    float* Bsel  = (float*)(ws + (size_t)16 * 1024 * 1024);               // 256 MiB
    float* Csel  = (float*)(ws + (size_t)(16 + 256) * 1024 * 1024);       // 256 MiB

    dim3 blk(256);
    // delta = softplus(x @ W_delta^T + bias)
    ssm_proj_gemm<true ><<<dim3(DIM / NT,            MROWS / MT), blk, 0, stream>>>(
        x, W_delta, d_bias, delta, DIM);
    // B_sel = x @ W_B^T ; C_sel = x @ W_C^T
    ssm_proj_gemm<false><<<dim3((DIM * NST) / NT,    MROWS / MT), blk, 0, stream>>>(
        x, W_B, nullptr, Bsel, DIM * NST);
    ssm_proj_gemm<false><<<dim3((DIM * NST) / NT,    MROWS / MT), blk, 0, stream>>>(
        x, W_C, nullptr, Csel, DIM * NST);

    // sequential selective scan
    ssm_scan<<<(BSZ * DIM * NST) / 256, blk, 0, stream>>>(x, delta, Bsel, Csel, A, y);
}
